// OptimizedQuantumLayer_11708080849239
// MI455X (gfx1250) — matrix operand order repaired and checked by reading, not compiled, not run on hardware
//
#include <hip/hip_runtime.h>
#include <hip/hip_bf16.h>
#include <math.h>

typedef __attribute__((ext_vector_type(16))) _Float16 v16h;
typedef __attribute__((ext_vector_type(8)))  float    v8f;
typedef __attribute__((ext_vector_type(4)))  float    v4f;

#define PI_F 3.14159265358979323846f

// ---------------------------------------------------------------------------
// Gate table for the fixed 23-gate unitary.
// kind: 0=rot2(A,B,th) 1=on_q0 rot1(P,th) 2=on_q1 rot1(P,th) 3=CX 4=CZ
// Pauli index: 0=X 1=Y 2=Z.  src: 0=theta 1=phi 2=ent 3=none
// ---------------------------------------------------------------------------
__constant__ int GK[23] = {0,0,0,0,0,0, 1,2,1,2,1,2,1,2, 3, 1, 4, 2, 0,0,0,0,0};
__constant__ int GA[23] = {2,0,1,2,1,2, 2,0,1,2,0,1,2,0, 0, 1, 0, 2, 0,1,2,1,2};
__constant__ int GB[23] = {2,0,1,0,2,2, 0,0,0,0,0,0,0,0, 0, 0, 0, 0, 0,1,0,2,2};
__constant__ int GS[23] = {1,1,1,1,1,1, 0,0,0,0,0,0,0,0, 3, 2, 3, 2, 2,2,2,2,2};
__constant__ int GI[23] = {0,1,2,3,4,5, 4,5,6,7,8,9,10,11, 0, 0, 0, 1, 2,3,4,5,6};

__constant__ float PRe[3][2][2] = {
  {{0.f, 1.f}, {1.f,  0.f}},   // X
  {{0.f, 0.f}, {0.f,  0.f}},   // Y
  {{1.f, 0.f}, {0.f, -1.f}},   // Z
};
__constant__ float PIm[3][2][2] = {
  {{0.f,  0.f}, {0.f, 0.f}},
  {{0.f, -1.f}, {1.f, 0.f}},
  {{0.f,  0.f}, {0.f, 0.f}},
};

// One wave builds the 4x4 complex unitary U into d_ws (interleaved re,im).
__global__ void build_unitary_kernel(const float* __restrict__ theta,
                                     const float* __restrict__ phi,
                                     const float* __restrict__ ent,
                                     float* __restrict__ Uout) {
  __shared__ float Ur[16], Ui[16];
  const int e = threadIdx.x;
  const int r = e >> 2, c = e & 3;
  if (e < 16) { Ur[e] = (r == c) ? 1.f : 0.f; Ui[e] = 0.f; }
  __syncthreads();

  for (int g = 0; g < 23; ++g) {
    float gr[4] = {0.f, 0.f, 0.f, 0.f};
    float gi[4] = {0.f, 0.f, 0.f, 0.f};
    if (e < 16) {
      const int kind = GK[g];
      float ang = 0.f;
      const int s = GS[g];
      if (s == 0) ang = theta[GI[g]];
      else if (s == 1) ang = phi[GI[g]];
      else if (s == 2) ang = ent[GI[g]];
      float sh, ch;
      __sincosf(0.5f * ang, &sh, &ch);
      for (int k = 0; k < 4; ++k) {
        float greal = 0.f, gimag = 0.f;
        if (kind == 0) {
          const int A = GA[g], Bp = GB[g];
          const float br = PRe[Bp][r >> 1][k >> 1], bi = PIm[Bp][r >> 1][k >> 1];
          const float ar = PRe[A][r & 1][k & 1],    ai = PIm[A][r & 1][k & 1];
          const float kr = br * ar - bi * ai;
          const float ki = br * ai + bi * ar;
          greal = ((r == k) ? ch : 0.f) + sh * ki;   // cos*I - i*sin*K
          gimag = -sh * kr;
        } else if (kind == 1) {           // kron(I2, g2)
          if ((r >> 1) == (k >> 1)) {
            const int P = GA[g];
            greal = (((r & 1) == (k & 1)) ? ch : 0.f) + sh * PIm[P][r & 1][k & 1];
            gimag = -sh * PRe[P][r & 1][k & 1];
          }
        } else if (kind == 2) {           // kron(g2, I2)
          if ((r & 1) == (k & 1)) {
            const int P = GA[g];
            greal = (((r >> 1) == (k >> 1)) ? ch : 0.f) + sh * PIm[P][r >> 1][k >> 1];
            gimag = -sh * PRe[P][r >> 1][k >> 1];
          }
        } else if (kind == 3) {           // CX little-endian
          const int col = (r == 0) ? 0 : (r == 1) ? 3 : (r == 2) ? 2 : 1;
          greal = (k == col) ? 1.f : 0.f;
        } else {                          // CZ
          greal = (r == k) ? ((r == 3) ? -1.f : 1.f) : 0.f;
        }
        gr[k] = greal; gi[k] = gimag;
      }
    }
    __syncthreads();
    float nr = 0.f, ni = 0.f;
    if (e < 16) {
      for (int k = 0; k < 4; ++k) {
        const float ur = Ur[k * 4 + c], ui = Ui[k * 4 + c];
        nr += gr[k] * ur - gi[k] * ui;
        ni += gr[k] * ui + gi[k] * ur;
      }
    }
    __syncthreads();
    if (e < 16) { Ur[e] = nr; Ui[e] = ni; }
    __syncthreads();
  }
  if (e < 16) { Uout[2 * e] = Ur[e]; Uout[2 * e + 1] = Ui[e]; }
}

// ---------------------------------------------------------------------------
// Fused: z = x@w (WMMA f16->f32), then per-row 2-qubit circuit + mitigation.
// 256 threads = 8 waves; each wave owns a 16-row tile; block owns 128 rows.
// ---------------------------------------------------------------------------
__global__ __launch_bounds__(256) void fused_qlayer_kernel(
    const float* __restrict__ x, const float* __restrict__ w,
    const float* __restrict__ bias, const float* __restrict__ theta,
    const float* __restrict__ Uws, float* __restrict__ out, int D) {
  __shared__ _Float16 ldsb[16384];        // B fragments: [D/32][32 lanes][16 halves]
  __shared__ float zbuf[8][16][2];
  __shared__ float Ur[16], Ui[16];

  const int tid  = threadIdx.x;
  const int lane = tid & 31;
  const int wv   = tid >> 5;
  const int iters = D >> 5;               // D/32 (== 32 for D=1024)

  // Stage w into the 16-bit B-matrix (32x16 KxN) WMMA fragment layout.
  for (int idx = tid; idx < iters * 512; idx += 256) {
    const int h  = idx & 15;
    const int ln = (idx >> 4) & 31;
    const int it = idx >> 9;
    const int K  = it * 32 + ((h >> 3) << 4) + ((ln >> 4) << 3) + (h & 7);
    const int N  = ln & 15;
    const float v = (N < 2) ? w[K * 2 + N] : 0.f;
    ldsb[idx] = (_Float16)v;
  }
  if (tid < 16) { Ur[tid] = Uws[2 * tid]; Ui[tid] = Uws[2 * tid + 1]; }
  __syncthreads();

  // ---- GEMM: 16 rows per wave, K in chunks of 32 ----
  const int rowbase = blockIdx.x * 128 + wv * 16;
  const int row     = rowbase + (lane & 15);
  const float* xr   = x + (size_t)row * D;
  const int koff    = (lane >> 4) ? 8 : 0;   // A layout: low lanes K0-7/16-23, high lanes K8-15/24-31

  v8f acc = {};
  for (int it = 0; it < iters; ++it) {
    const int k0 = it * 32 + koff;
    const v4f f0 = __builtin_nontemporal_load((const v4f*)(xr + k0));
    const v4f f1 = __builtin_nontemporal_load((const v4f*)(xr + k0 + 4));
    const v4f f2 = __builtin_nontemporal_load((const v4f*)(xr + k0 + 16));
    const v4f f3 = __builtin_nontemporal_load((const v4f*)(xr + k0 + 20));

    v16h a;
    a[0]  = (_Float16)f0.x; a[1]  = (_Float16)f0.y; a[2]  = (_Float16)f0.z; a[3]  = (_Float16)f0.w;
    a[4]  = (_Float16)f1.x; a[5]  = (_Float16)f1.y; a[6]  = (_Float16)f1.z; a[7]  = (_Float16)f1.w;
    a[8]  = (_Float16)f2.x; a[9]  = (_Float16)f2.y; a[10] = (_Float16)f2.z; a[11] = (_Float16)f2.w;
    a[12] = (_Float16)f3.x; a[13] = (_Float16)f3.y; a[14] = (_Float16)f3.z; a[15] = (_Float16)f3.w;

    const v16h b = *(const v16h*)&ldsb[(it * 32 + lane) * 16];

    acc = __builtin_amdgcn_wmma_f32_16x16x32_f16(
        /*neg_a=*/false, a, /*neg_b=*/false, b,
        /*c_mod=*/(short)0, acc, /*reuse_a=*/false, /*reuse_b=*/false);
  }

  // D tile: lane = N (mod 16), VGPR v = M (low lanes M=v, high lanes M=v+8).
  // We only need N=0,1 -> lanes 0,1,16,17 hold all 16 rows' z0/z1.
  if ((lane & 14) == 0) {
    const int n    = lane & 1;
    const int half = lane >> 4;
    #pragma unroll
    for (int v = 0; v < 8; ++v) zbuf[wv][half * 8 + v][n] = acc[v];
  }
  __syncthreads();

  // ---- per-row quantum circuit + error mitigation (threads 0..127) ----
  if (tid < 128) {
    const float b0 = bias[0], b1 = bias[1];
    const float th0 = theta[0], th1 = theta[1], th2 = theta[2], th3 = theta[3];
    const int wsrc = tid >> 4, rr = tid & 15;

    float z0 = zbuf[wsrc][rr][0] + b0;
    float z1 = zbuf[wsrc][rr][1] + b1;
    z0 = fminf(fmaxf(z0, -PI_F), PI_F);
    z1 = fminf(fmaxf(z1, -PI_F), PI_F);
    const float qa = tanhf(z0) * PI_F;
    const float qb = tanhf(z1) * PI_F;
    const float a0 = qa * th0, a1 = qb * th1, a2 = qa * th2, a3 = qb * th3;

    float s0, c0, s1, c1, s2, c2, s3, c3;
    __sincosf(0.5f * a0, &s0, &c0);
    __sincosf(0.5f * a1, &s1, &c1);
    __sincosf(0.5f * a2, &s2, &c2);
    __sincosf(0.5f * a3, &s3, &c3);

    // q0s = [c0*e^{-i a1/2}, s0*e^{+i a1/2}]
    const float q0r0 = c0 * c1, q0i0 = -c0 * s1;
    const float q0r1 = s0 * c1, q0i1 =  s0 * s1;
    // q1s = [c3*c2 + i s3*s2, s3*c2 - i c3*s2]
    const float q1r0 = c3 * c2, q1i0 =  s3 * s2;
    const float q1r1 = s3 * c2, q1i1 = -c3 * s2;

    // psi[2*j1 + i0] = q1s[j1] * q0s[i0]
    float pr[4], pim[4];
    pr[0] = q1r0 * q0r0 - q1i0 * q0i0;  pim[0] = q1r0 * q0i0 + q1i0 * q0r0;
    pr[1] = q1r0 * q0r1 - q1i0 * q0i1;  pim[1] = q1r0 * q0i1 + q1i0 * q0r1;
    pr[2] = q1r1 * q0r0 - q1i1 * q0i0;  pim[2] = q1r1 * q0i0 + q1i1 * q0r0;
    pr[3] = q1r1 * q0r1 - q1i1 * q0i1;  pim[3] = q1r1 * q0i1 + q1i1 * q0r1;

    // Only rows 0 and 3 of U matter (p00, p11).
    float o0r = 0.f, o0i = 0.f, o3r = 0.f, o3i = 0.f;
    #pragma unroll
    for (int j = 0; j < 4; ++j) {
      const float u0r = Ur[j],      u0i = Ui[j];
      const float u3r = Ur[12 + j], u3i = Ui[12 + j];
      o0r += u0r * pr[j] - u0i * pim[j];  o0i += u0r * pim[j] + u0i * pr[j];
      o3r += u3r * pr[j] - u3i * pim[j];  o3i += u3r * pim[j] + u3i * pr[j];
    }
    const float p00 = o0r * o0r + o0i * o0i;
    const float p11 = o3r * o3r + o3i * o3i;

    const float MITC = 0.99f * expf(-1.f / 100.f) * expf(-1.f / 50.f);
    const float m0 = fminf(fmaxf((p00 - 0.01f) * (1.f / 0.98f) * MITC, 0.f), 1.f);
    const float m1 = fminf(fmaxf((p11 - 0.01f) * (1.f / 0.98f) * MITC, 0.f), 1.f);
    const float tot = m0 + m1;
    const float inv = 1.f / fmaxf(tot, 1e-30f);
    const float o0 = (tot > 0.f) ? m0 * inv : 0.5f;
    const float o1 = (tot > 0.f) ? m1 * inv : 0.5f;

    const size_t orow = (size_t)blockIdx.x * 128 + tid;
    out[orow * 2 + 0] = o0;
    out[orow * 2 + 1] = o1;
  }
}

extern "C" void kernel_launch(void* const* d_in, const int* in_sizes, int n_in,
                              void* d_out, int out_size, void* d_ws, size_t ws_size,
                              hipStream_t stream) {
  const float* x     = (const float*)d_in[0];
  const float* w     = (const float*)d_in[1];
  const float* bias  = (const float*)d_in[2];
  const float* theta = (const float*)d_in[3];
  const float* phi   = (const float*)d_in[4];
  const float* ent   = (const float*)d_in[5];
  float* out = (float*)d_out;
  float* Uws = (float*)d_ws;

  const int D = in_sizes[1] / 2;      // 1024
  const int B = in_sizes[0] / D;      // 65536

  build_unitary_kernel<<<1, 32, 0, stream>>>(theta, phi, ent, Uws);
  fused_qlayer_kernel<<<B / 128, 256, 0, stream>>>(x, w, bias, theta, Uws, out, D);
}
